// _SelectiveSSMMixer_90228672955121
// MI455X (gfx1250) — matrix operand order repaired and checked by reading, not compiled, not run on hardware
//
#include <hip/hip_runtime.h>
#include <hip/hip_bf16.h>

// ---------- types ----------
typedef __bf16 bf16_t;
typedef __attribute__((ext_vector_type(16))) __bf16 v16bf;
typedef __attribute__((ext_vector_type(8)))  __bf16 v8bf;
typedef __attribute__((ext_vector_type(8)))  float   v8f;
typedef __attribute__((ext_vector_type(4)))  int     v4i;

// ---------- problem constants ----------
#define MODEL_DIM 1024
#define INNER_DIM 2048
#define NSTATE    16
#define BATCH     2
#define SEQLEN    1024
#define NROWS     (BATCH * SEQLEN)     // 2048 GEMM rows
#define PROJW     (2 * INNER_DIM)      // 4096

// ---------- CDNA5 async global->LDS copy (ASYNCcnt path) ----------
#if defined(__gfx1250__) && __has_builtin(__builtin_amdgcn_global_load_async_to_lds_b128)
#define HAVE_ASYNC 1
#else
#define HAVE_ASYNC 0
#endif

#if HAVE_ASYNC
typedef __attribute__((address_space(1))) v4i* as1_v4i_p;
typedef __attribute__((address_space(3))) v4i* as3_v4i_p;
#endif

__device__ __forceinline__ void cp16(const bf16_t* g, bf16_t* l) {
#if HAVE_ASYNC
    __builtin_amdgcn_global_load_async_to_lds_b128(
        (as1_v4i_p)(unsigned long long)g,
        (as3_v4i_p)(unsigned int)(unsigned long long)l,
        0, 0);
#else
    *(v8bf*)l = *(const v8bf*)g;      // global_load_b128 + ds_store_b128
#endif
}

__device__ __forceinline__ void sync_tiles() {
#if HAVE_ASYNC
#if __has_builtin(__builtin_amdgcn_s_wait_asynccnt)
    __builtin_amdgcn_s_wait_asynccnt(0);
#else
    asm volatile("s_wait_asynccnt 0" ::: "memory");
#endif
#endif
    __syncthreads();
}

// =======================================================================
// fp32 -> bf16 cast (grid-stride)
// =======================================================================
__global__ void cast_f32_bf16(const float* __restrict__ s, bf16_t* __restrict__ d, long n) {
    long i = (long)blockIdx.x * blockDim.x + threadIdx.x;
    long stride = (long)gridDim.x * blockDim.x;
    for (; i < n; i += stride) d[i] = (bf16_t)s[i];
}

// =======================================================================
// WMMA bf16 GEMM:  C[M,N] = A[M,K] @ W[N,K]^T   (fp32 accumulate)
// Block tile 128x128, 256 threads = 8 waves, wave tile 32(M) x 64(N).
// Double-buffered LDS staging (async global->LDS when available).
// Fragment layouts per CDNA5 ISA 7.12.2 (16-bit A 16x32; B 32x16; f32 C/D).
// =======================================================================
#define TILE_K 32
#define LPITCH 40   // 32-elem K row padded by 8 -> 80B pitch, conflict-free b128 reads

__device__ __forceinline__ v16bf lds_a_frag(const bf16_t* sA, int row, int lane) {
    int lm = lane & 15, lh = lane >> 4;
    const bf16_t* p = sA + (row + lm) * LPITCH + lh * 8;
    union { v16bf v; v8bf h[2]; } u;
    u.h[0] = *(const v8bf*)p;          // K = lh*8      .. +7
    u.h[1] = *(const v8bf*)(p + 16);   // K = 16 + lh*8 .. +7
    return u.v;
}

__device__ __forceinline__ v16bf lds_b_frag(const bf16_t* sB, int col, int lane) {
    int lm = lane & 15, lh = lane >> 4;
    // lanes 0-15: K=0..7,16..23 of column (col+lm); lanes 16-31: K=8..15,24..31
    const bf16_t* p = sB + (col + lm) * LPITCH + lh * 8;
    union { v16bf v; v8bf h[2]; } u;
    u.h[0] = *(const v8bf*)p;
    u.h[1] = *(const v8bf*)(p + 16);
    return u.v;
}

__global__ __launch_bounds__(256) void gemm_bf16_nt(const bf16_t* __restrict__ A,
                                                    const bf16_t* __restrict__ W,
                                                    float* __restrict__ C,
                                                    int M, int N, int K) {
    __shared__ bf16_t sA[2][128 * LPITCH];
    __shared__ bf16_t sB[2][128 * LPITCH];

    const int tid  = threadIdx.x;
    const int lane = tid & 31;
    const int wave = tid >> 5;                 // 0..7
    const int wm   = wave & 3;                 // 4 wave-rows
    const int wn   = wave >> 2;                // 2 wave-cols
    const int bm   = blockIdx.y * 128;
    const int bn   = blockIdx.x * 128;

    // cooperative copy: each thread moves 4x16B per K-step (2 for A, 2 for B)
    const int crow = tid >> 2;                 // 0..63
    const int cc   = (tid & 3) * 8;            // element offset within 32-wide K slab

    auto copy_tiles = [&](int buf, int k) {
        cp16(A + (size_t)(bm + crow)      * K + k + cc, &sA[buf][(crow)      * LPITCH + cc]);
        cp16(A + (size_t)(bm + 64 + crow) * K + k + cc, &sA[buf][(64 + crow) * LPITCH + cc]);
        cp16(W + (size_t)(bn + crow)      * K + k + cc, &sB[buf][(crow)      * LPITCH + cc]);
        cp16(W + (size_t)(bn + 64 + crow) * K + k + cc, &sB[buf][(64 + crow) * LPITCH + cc]);
    };

    v8f acc[2][4] = {};
    const int nk = K / TILE_K;

    copy_tiles(0, 0);
    sync_tiles();

    for (int kk = 0; kk < nk; ++kk) {
        const int buf = kk & 1;
        if (kk + 1 < nk) copy_tiles(buf ^ 1, (kk + 1) * TILE_K);

        v16bf a0 = lds_a_frag(sA[buf], wm * 32,      lane);
        v16bf a1 = lds_a_frag(sA[buf], wm * 32 + 16, lane);
        v16bf b0 = lds_b_frag(sB[buf], wn * 64,      lane);
        v16bf b1 = lds_b_frag(sB[buf], wn * 64 + 16, lane);
        v16bf b2 = lds_b_frag(sB[buf], wn * 64 + 32, lane);
        v16bf b3 = lds_b_frag(sB[buf], wn * 64 + 48, lane);

        acc[0][0] = __builtin_amdgcn_wmma_f32_16x16x32_bf16(false, a0, false, b0, (short)0, acc[0][0], false, false);
        acc[0][1] = __builtin_amdgcn_wmma_f32_16x16x32_bf16(false, a0, false, b1, (short)0, acc[0][1], false, false);
        acc[0][2] = __builtin_amdgcn_wmma_f32_16x16x32_bf16(false, a0, false, b2, (short)0, acc[0][2], false, false);
        acc[0][3] = __builtin_amdgcn_wmma_f32_16x16x32_bf16(false, a0, false, b3, (short)0, acc[0][3], false, false);
        acc[1][0] = __builtin_amdgcn_wmma_f32_16x16x32_bf16(false, a1, false, b0, (short)0, acc[1][0], false, false);
        acc[1][1] = __builtin_amdgcn_wmma_f32_16x16x32_bf16(false, a1, false, b1, (short)0, acc[1][1], false, false);
        acc[1][2] = __builtin_amdgcn_wmma_f32_16x16x32_bf16(false, a1, false, b2, (short)0, acc[1][2], false, false);
        acc[1][3] = __builtin_amdgcn_wmma_f32_16x16x32_bf16(false, a1, false, b3, (short)0, acc[1][3], false, false);

        if (kk + 1 < nk) sync_tiles();
    }

    // C/D layout: VGPR r, lane l -> M = r + 8*(l>>4), N = l&15
    const int lm = lane & 15, lh = lane >> 4;
    const int rowBase = bm + wm * 32;
    const int colBase = bn + wn * 64;
    #pragma unroll
    for (int i = 0; i < 2; ++i)
        #pragma unroll
        for (int j = 0; j < 4; ++j)
            #pragma unroll
            for (int r = 0; r < 8; ++r)
                C[(size_t)(rowBase + i * 16 + r + 8 * lh) * N + colBase + j * 16 + lm] = acc[i][j][r];
}

// =======================================================================
// depthwise causal conv (K=4) + bias + SiLU on the first INNER cols of proj
// =======================================================================
__global__ void conv_silu(const float* __restrict__ proj,
                          const float* __restrict__ wconv,  // [INNER,1,4]
                          const float* __restrict__ bconv,  // [INNER]
                          float* __restrict__ u_f32, bf16_t* __restrict__ u_bf) {
    long idx = (long)blockIdx.x * blockDim.x + threadIdx.x;   // < NROWS*INNER
    if (idx >= (long)NROWS * INNER_DIM) return;
    int c  = (int)(idx % INNER_DIM);
    long bs = idx / INNER_DIM;
    int s  = (int)(bs % SEQLEN);
    long rowb = bs - s;                                       // b*SEQLEN
    float acc = bconv[c];
    #pragma unroll
    for (int j = 0; j < 4; ++j) {
        int t = s - 3 + j;
        if (t >= 0)
            acc += wconv[c * 4 + j] * proj[(rowb + t) * (long)PROJW + c];
    }
    float sv = acc / (1.0f + __expf(-acc));                   // SiLU
    u_f32[idx] = sv;
    u_bf[idx]  = (bf16_t)sv;
}

// =======================================================================
// delta = softplus(delta_raw + b_dt[c])  (in place)
// =======================================================================
__global__ void softplus_bias(float* __restrict__ delta, const float* __restrict__ b_dt) {
    long idx = (long)blockIdx.x * blockDim.x + threadIdx.x;
    if (idx >= (long)NROWS * INNER_DIM) return;
    int c = (int)(idx % INNER_DIM);
    float x = delta[idx] + b_dt[c];
    delta[idx] = (x > 20.0f) ? x : log1pf(__expf(x));
}

// =======================================================================
// Sequential selective scan: one lane per (batch, channel).
// state_n <- exp(dt*a_n)*state_n + dt*u*b_n ;  y = sum c_n*state_n + d*u
// out = y * sigmoid(gate);  gate lives in proj[:, INNER + c].
// =======================================================================
__global__ __launch_bounds__(256) void ssm_scan(const float* __restrict__ delta,
                                                const float* __restrict__ u,
                                                const float* __restrict__ proj,
                                                const float* __restrict__ bmat,
                                                const float* __restrict__ cmat,
                                                const float* __restrict__ a_log,
                                                const float* __restrict__ dvec,
                                                bf16_t* __restrict__ mixed_bf) {
    int ch = blockIdx.x * blockDim.x + threadIdx.x;     // 0 .. BATCH*INNER-1
    if (ch >= BATCH * INNER_DIM) return;
    int b = ch >> 11;                                   // / INNER_DIM
    int c = ch & (INNER_DIM - 1);

    float an[NSTATE];
    #pragma unroll
    for (int n = 0; n < NSTATE; ++n) an[n] = -__expf(a_log[c * NSTATE + n]);
    float st[NSTATE];
    #pragma unroll
    for (int n = 0; n < NSTATE; ++n) st[n] = 0.0f;
    const float dd = dvec[c];

    for (int t = 0; t < SEQLEN; ++t) {
        long row = (long)b * SEQLEN + t;
        float dt = delta[row * INNER_DIM + c];
        float ut = u[row * INNER_DIM + c];
        float du = dt * ut;
        const float4* bp = (const float4*)(bmat + row * (long)(INNER_DIM * NSTATE) + (long)c * NSTATE);
        const float4* cp = (const float4*)(cmat + row * (long)(INNER_DIM * NSTATE) + (long)c * NSTATE);
        float y = 0.0f;
        #pragma unroll
        for (int q = 0; q < 4; ++q) {
            float4 bv = bp[q];
            float4 cv = cp[q];
            int n = q * 4;
            st[n+0] = __expf(dt * an[n+0]) * st[n+0] + du * bv.x;  y += cv.x * st[n+0];
            st[n+1] = __expf(dt * an[n+1]) * st[n+1] + du * bv.y;  y += cv.y * st[n+1];
            st[n+2] = __expf(dt * an[n+2]) * st[n+2] + du * bv.z;  y += cv.z * st[n+2];
            st[n+3] = __expf(dt * an[n+3]) * st[n+3] + du * bv.w;  y += cv.w * st[n+3];
        }
        y += dd * ut;
        float gr = proj[row * (long)PROJW + INNER_DIM + c];
        float g  = 1.0f / (1.0f + __expf(-gr));
        mixed_bf[row * INNER_DIM + c] = (bf16_t)(y * g);
    }
}

// =======================================================================
// host-side launcher
// =======================================================================
extern "C" void kernel_launch(void* const* d_in, const int* in_sizes, int n_in,
                              void* d_out, int out_size, void* d_ws, size_t ws_size,
                              hipStream_t stream) {
    (void)in_sizes; (void)n_in; (void)out_size; (void)ws_size;
    const float* hidden = (const float*)d_in[0];   // [2,1024,1024]
    const float* w_in   = (const float*)d_in[1];   // [4096,1024]
    const float* w_conv = (const float*)d_in[2];   // [2048,1,4]
    const float* b_conv = (const float*)d_in[3];   // [2048]
    const float* w_dt   = (const float*)d_in[4];   // [2048,2048]
    const float* b_dt   = (const float*)d_in[5];   // [2048]
    const float* w_b    = (const float*)d_in[6];   // [32768,2048]
    const float* w_c    = (const float*)d_in[7];   // [32768,2048]
    const float* w_out  = (const float*)d_in[8];   // [1024,2048]
    const float* a_log  = (const float*)d_in[9];   // [2048,16]
    const float* dvec   = (const float*)d_in[10];  // [2048]
    float* out = (float*)d_out;                    // [2,1024,1024]

    // ---- workspace carve-out ----
    char* w = (char*)d_ws;
    auto carve = [&](size_t bytes) -> void* {
        void* p = (void*)w;
        w += (bytes + 255) & ~(size_t)255;
        return p;
    };
    const long NH   = (long)NROWS * MODEL_DIM;              // 2M
    const long NWIN = (long)PROJW * MODEL_DIM;              // 4M
    const long NWDT = (long)INNER_DIM * INNER_DIM;          // 4M
    const long NWB  = (long)INNER_DIM * NSTATE * INNER_DIM; // 67.1M
    const long NWO  = (long)MODEL_DIM * INNER_DIM;          // 2M
    const long NU   = (long)NROWS * INNER_DIM;              // 4M
    const long NBC  = (long)NROWS * INNER_DIM * NSTATE;     // 67.1M

    bf16_t* hid_bf  = (bf16_t*)carve(NH   * 2);
    bf16_t* win_bf  = (bf16_t*)carve(NWIN * 2);
    bf16_t* wdt_bf  = (bf16_t*)carve(NWDT * 2);
    bf16_t* wb_bf   = (bf16_t*)carve(NWB  * 2);
    bf16_t* wc_bf   = (bf16_t*)carve(NWB  * 2);
    bf16_t* wout_bf = (bf16_t*)carve(NWO  * 2);
    float*  proj    = (float*) carve((long)NROWS * PROJW * 4);
    float*  u_f32   = (float*) carve(NU * 4);
    bf16_t* u_bf    = (bf16_t*)carve(NU * 2);
    float*  delta   = (float*) carve(NU * 4);
    float*  bmat    = (float*) carve(NBC * 4);
    float*  cmat    = (float*) carve(NBC * 4);
    bf16_t* mix_bf  = (bf16_t*)carve(NU * 2);

    // ---- 1. casts ----
    cast_f32_bf16<<<2048, 256, 0, stream>>>(hidden, hid_bf, NH);
    cast_f32_bf16<<<2048, 256, 0, stream>>>(w_in,   win_bf, NWIN);
    cast_f32_bf16<<<2048, 256, 0, stream>>>(w_dt,   wdt_bf, NWDT);
    cast_f32_bf16<<<8192, 256, 0, stream>>>(w_b,    wb_bf,  NWB);
    cast_f32_bf16<<<8192, 256, 0, stream>>>(w_c,    wc_bf,  NWB);
    cast_f32_bf16<<<2048, 256, 0, stream>>>(w_out,  wout_bf, NWO);

    // ---- 2. proj = hidden @ w_in^T  [2048 x 4096] ----
    gemm_bf16_nt<<<dim3(PROJW / 128, NROWS / 128), 256, 0, stream>>>(
        hid_bf, win_bf, proj, NROWS, PROJW, MODEL_DIM);

    // ---- 3. depthwise conv + SiLU -> u ----
    {
        long total = (long)NROWS * INNER_DIM;
        conv_silu<<<(unsigned)((total + 255) / 256), 256, 0, stream>>>(
            proj, w_conv, b_conv, u_f32, u_bf);
    }

    // ---- 4. delta_raw = u @ w_dt^T ; softplus(+b_dt) ----
    gemm_bf16_nt<<<dim3(INNER_DIM / 128, NROWS / 128), 256, 0, stream>>>(
        u_bf, wdt_bf, delta, NROWS, INNER_DIM, INNER_DIM);
    {
        long total = (long)NROWS * INNER_DIM;
        softplus_bias<<<(unsigned)((total + 255) / 256), 256, 0, stream>>>(delta, b_dt);
    }

    // ---- 5. bmat / cmat = u @ w_{b,c}^T  [2048 x 32768] ----
    gemm_bf16_nt<<<dim3((INNER_DIM * NSTATE) / 128, NROWS / 128), 256, 0, stream>>>(
        u_bf, wb_bf, bmat, NROWS, INNER_DIM * NSTATE, INNER_DIM);
    gemm_bf16_nt<<<dim3((INNER_DIM * NSTATE) / 128, NROWS / 128), 256, 0, stream>>>(
        u_bf, wc_bf, cmat, NROWS, INNER_DIM * NSTATE, INNER_DIM);

    // ---- 6. sequential selective scan (+ gate) ----
    ssm_scan<<<(BATCH * INNER_DIM) / 256, 256, 0, stream>>>(
        delta, u_f32, proj, bmat, cmat, a_log, dvec, mix_bf);

    // ---- 7. out = mixed @ w_out^T  [2048 x 1024] ----
    gemm_bf16_nt<<<dim3(MODEL_DIM / 128, NROWS / 128), 256, 0, stream>>>(
        mix_bf, wout_bf, out, NROWS, MODEL_DIM, INNER_DIM);
}
